// ChannelCovarianceBlock_79706003079468
// MI455X (gfx1250) — matrix operand order repaired and checked by reading, not compiled, not run on hardware
//
#include <hip/hip_runtime.h>
#include <hip/hip_bf16.h>

typedef __attribute__((ext_vector_type(16))) __bf16 v16bf;
typedef __attribute__((ext_vector_type(8)))  float  v8f;

union ABu { uint4 q[2]; v16bf v; };

#define NW   10
#define BS   10
#define CH   128
#define HW   784      // 28*28
#define NN   1280     // BS*CH
#define BQ   256      // query batch
#define PADK 800      // HW padded to 25*32 = 10*80
#define KT5  25       // PADK/32
#define KT4  40       // NN/32
#define QG   10       // PADK/80 q-groups
#define MB   2        // M blocking: 2 x 16 rows
#define NB   5        // N blocking: 5 x 16 cols

__device__ __forceinline__ unsigned short f32_to_bf16(float f) {
    unsigned int u = __float_as_uint(f);
    unsigned int r = u + 0x7FFFu + ((u >> 16) & 1u);  // round-to-nearest-even
    return (unsigned short)(r >> 16);
}
__device__ __forceinline__ float bf16_to_f32(unsigned short h) {
    return __uint_as_float(((unsigned int)h) << 16);
}

// ---------------- K0: per-class column means of x2 ----------------
__global__ void k_colmean(const float* __restrict__ x2, float* __restrict__ mean) {
    int n = blockIdx.y;
    int p = blockIdx.x * 256 + threadIdx.x;
    if (p >= HW) return;
    const float* base = x2 + (size_t)n * NN * HW + p;
    float s = 0.f;
    for (int i = 0; i < NN; ++i) s += base[(size_t)i * HW];
    mean[n * HW + p] = s * (1.0f / NN);
}

// ---- K1: center + cast to bf16; Xc (1280 x 800, zero-pad cols) and
// ----     XcT (800 x 1280, zero-pad rows) via LDS tile transpose ----
__global__ void k_center(const float* __restrict__ x2, const float* __restrict__ mean,
                         unsigned short* __restrict__ Xc, unsigned short* __restrict__ XcT) {
    __shared__ float tile[32][33];
    int n  = blockIdx.z;
    int i0 = blockIdx.y * 32;
    int p0 = blockIdx.x * 32;        // p0 <= 768, covers p up to 799
    int tx = threadIdx.x, ty = threadIdx.y;
#pragma unroll
    for (int j = 0; j < 4; ++j) {
        int i = i0 + ty + j * 8;
        int p = p0 + tx;
        float v = 0.f;
        if (p < HW) v = x2[((size_t)n * NN + i) * HW + p] - mean[n * HW + p];
        Xc[((size_t)n * NN + i) * PADK + p] = f32_to_bf16(v);
        tile[ty + j * 8][tx] = v;
    }
    __syncthreads();
#pragma unroll
    for (int j = 0; j < 4; ++j) {
        int p = p0 + ty + j * 8;
        int i = i0 + tx;
        XcT[((size_t)n * PADK + p) * NN + i] = f32_to_bf16(tile[tx][ty + j * 8]);
    }
}

// ---------------- K2: diff = normalize+center x1 rows, bf16, K padded to 800 --
__global__ void k_diff(const float* __restrict__ x1, unsigned short* __restrict__ diffu) {
    int r    = blockIdx.x * 8 + (threadIdx.x >> 5);   // row = b*128 + c
    int lane = threadIdx.x & 31;
    const float* base = x1 + (size_t)r * HW;
    float s = 0.f, ss = 0.f;
    for (int p = lane; p < HW; p += 32) { float v = base[p]; s += v; ss += v * v; }
#pragma unroll
    for (int m = 16; m >= 1; m >>= 1) { s += __shfl_xor(s, m, 32); ss += __shfl_xor(ss, m, 32); }
    float inv = rsqrtf(ss);
    float mu  = s * inv * (1.0f / HW);
    unsigned short* drow = diffu + (size_t)r * PADK;
    for (int p = lane; p < HW; p += 32) drow[p] = f32_to_bf16(base[p] * inv - mu);
    if (lane < 16) drow[HW + lane] = 0;               // zero K-pad
}

// ---- K4: cov[n] = XcT(800x1280-rows) x Xc(1280x800) / 1279, 32x80 blocked ----
__global__ void __launch_bounds__(32, 1)
k_cov(const unsigned short* __restrict__ XcT,
      const unsigned short* __restrict__ Xc,
      unsigned short* __restrict__ covu) {
    int n  = blockIdx.z;
    int p0 = blockIdx.y * (MB * 16);   // 25 groups of 32 rows
    int q0 = blockIdx.x * (NB * 16);   // 10 groups of 80 cols
    int l = threadIdx.x;
    int mrow = l & 15;
    int half = l >> 4;

    const unsigned short* arow0 = XcT + ((size_t)n * PADK + p0 + mrow) * NN + half * 8;
    const unsigned short* bbase = XcT + ((size_t)n * PADK + q0 + mrow) * NN + half * 8;

    v8f acc[MB][NB];
#pragma unroll
    for (int r = 0; r < MB; ++r)
#pragma unroll
        for (int j = 0; j < NB; ++j) acc[r][j] = v8f{0.f,0.f,0.f,0.f,0.f,0.f,0.f,0.f};

    for (int kt = 0; kt < KT4; ++kt) {
        ABu a[MB], b[NB];
#pragma unroll
        for (int r = 0; r < MB; ++r) {
            const unsigned short* ap = arow0 + (size_t)r * 16 * NN + kt * 32;
            a[r].q[0] = *(const uint4*)(ap);
            a[r].q[1] = *(const uint4*)(ap + 16);
        }
        const unsigned short* bp = bbase + kt * 32;
#pragma unroll
        for (int j = 0; j < NB; ++j) {
            b[j].q[0] = *(const uint4*)(bp + (size_t)j * 16 * NN);
            b[j].q[1] = *(const uint4*)(bp + (size_t)j * 16 * NN + 16);
        }
#pragma unroll
        for (int r = 0; r < MB; ++r)
#pragma unroll
            for (int j = 0; j < NB; ++j)
                acc[r][j] = __builtin_amdgcn_wmma_f32_16x16x32_bf16(
                    false, a[r].v, false, b[j].v, (short)0, acc[r][j], false, false);
    }
    const float sc = 1.0f / (NN - 1);
#pragma unroll
    for (int r = 0; r < MB; ++r)
#pragma unroll
        for (int j = 0; j < NB; ++j)
#pragma unroll
            for (int v = 0; v < 8; ++v)
                covu[((size_t)n * PADK + p0 + r * 16 + v + half * 8) * PADK
                     + q0 + j * 16 + mrow] = f32_to_bf16(acc[r][j][v] * sc);
}

// ---- K5: sim[b,n,c] = diag(diff_b cov_n diff_b^T), fused, 32x80 blocked ----
__global__ void __launch_bounds__(32, 1)
k_sim(const unsigned short* __restrict__ diffu,
      const unsigned short* __restrict__ covu,
      float* __restrict__ out) {
    int n   = blockIdx.x;
    int ct0 = blockIdx.y * (MB * 16);  // 4 groups of 32 channels
    int b   = blockIdx.z;
    int l = threadIdx.x;
    int mrow = l & 15;
    int half = l >> 4;

    const unsigned short* arow0 = diffu + ((size_t)b * CH + ct0 + mrow) * PADK + half * 8;
    const unsigned short* bbase = covu + (size_t)n * PADK * PADK;

    float racc[MB][8];
#pragma unroll
    for (int r = 0; r < MB; ++r)
#pragma unroll
        for (int v = 0; v < 8; ++v) racc[r][v] = 0.f;

    for (int qg = 0; qg < QG; ++qg) {
        int q0 = qg * (NB * 16);
        v8f acc[MB][NB];
#pragma unroll
        for (int r = 0; r < MB; ++r)
#pragma unroll
            for (int j = 0; j < NB; ++j) acc[r][j] = v8f{0.f,0.f,0.f,0.f,0.f,0.f,0.f,0.f};

        for (int kt = 0; kt < KT5; ++kt) {
            ABu a[MB], bb[NB];
#pragma unroll
            for (int r = 0; r < MB; ++r) {
                const unsigned short* ap = arow0 + (size_t)r * 16 * PADK + kt * 32;
                a[r].q[0] = *(const uint4*)(ap);
                a[r].q[1] = *(const uint4*)(ap + 16);
            }
            const unsigned short* bp = bbase + ((size_t)q0 + mrow) * PADK + half * 8 + kt * 32;
#pragma unroll
            for (int j = 0; j < NB; ++j) {
                bb[j].q[0] = *(const uint4*)(bp + (size_t)j * 16 * PADK);
                bb[j].q[1] = *(const uint4*)(bp + (size_t)j * 16 * PADK + 16);
            }
#pragma unroll
            for (int r = 0; r < MB; ++r)
#pragma unroll
                for (int j = 0; j < NB; ++j)
                    acc[r][j] = __builtin_amdgcn_wmma_f32_16x16x32_bf16(
                        false, a[r].v, false, bb[j].v, (short)0, acc[r][j], false, false);
        }
        // fused diagonal: acc[r][j][v] = T[c = ct0+r*16+v+half*8][q0+j*16+mrow]
#pragma unroll
        for (int r = 0; r < MB; ++r) {
            const unsigned short* dcol =
                diffu + ((size_t)b * CH + ct0 + r * 16 + half * 8) * PADK + q0 + mrow;
#pragma unroll
            for (int j = 0; j < NB; ++j)
#pragma unroll
                for (int v = 0; v < 8; ++v)
                    racc[r][v] += acc[r][j][v] * bf16_to_f32(dcol[(size_t)v * PADK + j * 16]);
        }
    }
    // reduce each row-partial across the 16 lanes of this half, then store
#pragma unroll
    for (int r = 0; r < MB; ++r)
#pragma unroll
        for (int v = 0; v < 8; ++v) {
            float s = racc[r][v];
            s += __shfl_xor(s, 1, 32);
            s += __shfl_xor(s, 2, 32);
            s += __shfl_xor(s, 4, 32);
            s += __shfl_xor(s, 8, 32);
            racc[r][v] = s;
        }
    if (mrow == 0) {
#pragma unroll
        for (int r = 0; r < MB; ++r)
#pragma unroll
            for (int v = 0; v < 8; ++v)
                out[(size_t)b * (NW * CH) + n * CH + ct0 + r * 16 + half * 8 + v] = racc[r][v];
    }
}

extern "C" void kernel_launch(void* const* d_in, const int* in_sizes, int n_in,
                              void* d_out, int out_size, void* d_ws, size_t ws_size,
                              hipStream_t stream) {
    const float* x1 = (const float*)d_in[0];   // (256,128,28,28)
    const float* x2 = (const float*)d_in[1];   // (10,10,128,28,28)
    float* out = (float*)d_out;                // (256, 1280)

    char* ws = (char*)d_ws;
    const size_t OFF_MEAN = 0;                                     // 10*784 f32
    const size_t OFF_XC   = 31488;                                 // 10*1280*800 bf16
    const size_t OFF_XCT  = OFF_XC  + (size_t)NW * NN * PADK * 2;  // 10*800*1280 bf16
    const size_t OFF_COV  = OFF_XCT + (size_t)NW * PADK * NN * 2;  // 10*800*800 bf16
    const size_t OFF_DIFF = OFF_COV + (size_t)NW * PADK * PADK * 2;// 256*128*800 bf16

    float*          mean  = (float*)(ws + OFF_MEAN);
    unsigned short* Xc    = (unsigned short*)(ws + OFF_XC);
    unsigned short* XcT   = (unsigned short*)(ws + OFF_XCT);
    unsigned short* covu  = (unsigned short*)(ws + OFF_COV);
    unsigned short* diffu = (unsigned short*)(ws + OFF_DIFF);

    // K0: column means per class
    k_colmean<<<dim3((HW + 255) / 256, NW), 256, 0, stream>>>(x2, mean);
    // K1: center + cast, produce Xc (padded cols) and XcT (padded rows)
    k_center<<<dim3(PADK / 32, NN / 32, NW), dim3(32, 8), 0, stream>>>(x2, mean, Xc, XcT);
    // K2: normalized/centered query rows in bf16 (K padded to 800)
    k_diff<<<(BQ * CH) / 8, 256, 0, stream>>>(x1, diffu);
    // K4: covariance GEMMs (WMMA bf16, f32 accumulate), 32x80 register blocked
    k_cov<<<dim3(QG, PADK / 32, NW), 32, 0, stream>>>(XcT, Xc, covu);
    // K5: fused diff @ cov @ diff^T diagonal (WMMA bf16), 32x80 register blocked
    k_sim<<<dim3(NW, CH / 32, BQ), 32, 0, stream>>>(diffu, covu, out);
    (void)in_sizes; (void)n_in; (void)out_size; (void)ws_size;
}
